// gcn_74552042324153
// MI455X (gfx1250) — matrix operand order repaired and checked by reading, not compiled, not run on hardware
//
#include <hip/hip_runtime.h>

#define Ndim 32
#define Tdim 300
#define Vdim 25
#define Cdim 64
#define Kdim 3
#define Fdim 128
#define TT   10
#define NCHUNK (Tdim / TT) // 30

typedef __attribute__((ext_vector_type(16))) _Float16 v16h;
typedef __attribute__((ext_vector_type(8)))  _Float16 v8h;
typedef __attribute__((ext_vector_type(8)))  float    v8f;
typedef __attribute__((ext_vector_type(4)))  float    f32x4;

static __device__ __forceinline__ v16h cat16(v8h lo, v8h hi) {
  v16h r;
#pragma unroll
  for (int i = 0; i < 8; ++i) { r[i] = lo[i]; r[i + 8] = hi[i]; }
  return r;
}

static __device__ __forceinline__ v8h cvt8(f32x4 a, f32x4 b) {
  v8h r;
#pragma unroll
  for (int i = 0; i < 4; ++i) { r[i] = (_Float16)a[i]; r[i + 4] = (_Float16)b[i]; }
  return r;
}

__global__ __launch_bounds__(256, 1)
void stgcn_fused(const float* __restrict__ x, const float* __restrict__ A,
                 const float* __restrict__ W, const float* __restrict__ b,
                 float* __restrict__ y) {
  // W as f16, column-major [out-col 384][cin 64] : 48 KB
  __shared__ alignas(16) _Float16 wlds[Kdim * Fdim][Cdim];
  // A_k^T zero-padded to 32x32, row-major [w][v] : 6 KB
  __shared__ alignas(16) _Float16 at2[Kdim][32][32];
  // per-wave H_k staging tile, col-major [n 16][v 32] : 8 KB
  __shared__ alignas(16) _Float16 hst[8][16][32];

  const int tid  = threadIdx.x;
  const int lane = tid & 31;
  const int wv   = tid >> 5;   // 0..7 : which 16-wide column tile of F
  const int nb   = blockIdx.x;
  const int n    = nb / NCHUNK;
  const int t0   = (nb % NCHUNK) * TT;
  const int nl   = lane & 15;  // N-col within tile / row selector
  const int kh   = lane >> 4;  // half selector

  // ---- stage W into LDS (f32 -> f16, transposed to col-major) ----
  for (int i = tid; i < Kdim * Fdim * Cdim; i += 256) {
    int d = i >> 6;   // out col 0..383
    int c = i & 63;   // cin
    wlds[d][c] = (_Float16)W[c * (Kdim * Fdim) + d];
  }
  // ---- stage zero-padded A_k^T ----
  for (int i = tid; i < Kdim * 32 * 32; i += 256) {
    int k = i >> 10;
    int w = (i >> 5) & 31;
    int v = i & 31;
    at2[k][w][v] = (w < Vdim && v < Vdim) ? (_Float16)A[(k * Vdim + v) * Vdim + w]
                                          : (_Float16)0.f;
  }
  __syncthreads();

  // ---- hoist t-invariant WMMA operands ----
  // stage-1 B operands (W slices): [k][kstep]
  v16h bop1[Kdim][2];
#pragma unroll
  for (int k = 0; k < Kdim; ++k)
#pragma unroll
    for (int ks = 0; ks < 2; ++ks) {
      const _Float16* p = &wlds[k * Fdim + wv * 16 + nl][ks * 32 + kh * 8];
      bop1[k][ks] = cat16(*(const v8h*)p, *(const v8h*)(p + 16));
    }
  // stage-2 A operands (A_k^T): [k][mtile]
  v16h aop2[Kdim][2];
#pragma unroll
  for (int k = 0; k < Kdim; ++k)
#pragma unroll
    for (int mt = 0; mt < 2; ++mt) {
      const _Float16* p0 = &at2[k][mt * 16 + nl][kh * 8];
      const _Float16* p1 = &at2[k][mt * 16 + nl][16 + kh * 8];
      aop2[k][mt] = cat16(*(const v8h*)p0, *(const v8h*)p1);
    }
  // bias for this wave's columns (enters as stage-1 C initializer)
  float bv[Kdim];
#pragma unroll
  for (int k = 0; k < Kdim; ++k) bv[k] = b[k * Fdim + wv * 16 + nl];

  for (int tt = 0; tt < TT; ++tt) {
    const int t = t0 + tt;
    const float* xrow = x + (size_t)(n * Tdim + t) * (Vdim * Cdim);

    // x A-operands [kstep][mtile], f32 global -> f16 regs
    v16h xop[2][2];
#pragma unroll
    for (int mt = 0; mt < 2; ++mt) {
      int v = mt * 16 + nl;
      if (v > Vdim - 1) v = Vdim - 1;  // clamp pad rows (masked later by at2 zeros)
      const float* xr = xrow + v * Cdim;
#pragma unroll
      for (int ks = 0; ks < 2; ++ks) {
        const float* p0 = xr + ks * 32 + kh * 8;
        const float* p1 = xr + ks * 32 + 16 + kh * 8;
        xop[ks][mt] = cat16(cvt8(*(const f32x4*)p0, *(const f32x4*)(p0 + 4)),
                            cvt8(*(const f32x4*)p1, *(const f32x4*)(p1 + 4)));
      }
    }

    v8f yacc[2] = {};
#pragma unroll
    for (int k = 0; k < Kdim; ++k) {
      // ---- stage 1: H_k = x @ W_k + b_k ----
      v8f hacc[2];
#pragma unroll
      for (int mt = 0; mt < 2; ++mt) {
#pragma unroll
        for (int i = 0; i < 8; ++i) hacc[mt][i] = bv[k];
#pragma unroll
        for (int ks = 0; ks < 2; ++ks)
          hacc[mt] = __builtin_amdgcn_wmma_f32_16x16x32_f16(
              false, xop[ks][mt], false, bop1[k][ks], (short)0, hacc[mt], false, false);
        // D-layout -> col-major LDS (f32 -> f16 pack, one b128 store)
        v8h hh;
#pragma unroll
        for (int i = 0; i < 8; ++i) hh[i] = (_Float16)hacc[mt][i];
        *(v8h*)&hst[wv][nl][mt * 16 + kh * 8] = hh;
      }
      asm volatile("s_wait_dscnt 0" ::: "memory");
      // ---- stage 2: Y += A_k^T @ H_k (H_k re-read in B layout) ----
      const _Float16* hp = &hst[wv][nl][kh * 8];
      v16h bop2 = cat16(*(const v8h*)hp, *(const v8h*)(hp + 16));
#pragma unroll
      for (int mt = 0; mt < 2; ++mt)
        yacc[mt] = __builtin_amdgcn_wmma_f32_16x16x32_f16(
            false, aop2[k][mt], false, bop2, (short)0, yacc[mt], false, false);
      asm volatile("" ::: "memory");
    }

    // ---- store Y (rows w < 25) ----
    float* yb = y + (size_t)(n * Tdim + t) * (Vdim * Fdim) + (wv * 16 + nl);
#pragma unroll
    for (int mt = 0; mt < 2; ++mt)
#pragma unroll
      for (int r = 0; r < 8; ++r) {
        int w = mt * 16 + kh * 8 + r;
        if (w < Vdim) yb[w * Fdim] = yacc[mt][r];
      }
  }
}

extern "C" void kernel_launch(void* const* d_in, const int* in_sizes, int n_in,
                              void* d_out, int out_size, void* d_ws, size_t ws_size,
                              hipStream_t stream) {
  (void)in_sizes; (void)n_in; (void)d_ws; (void)ws_size; (void)out_size;
  const float* x = (const float*)d_in[0];
  const float* A = (const float*)d_in[1];
  const float* W = (const float*)d_in[2];
  const float* b = (const float*)d_in[3];
  float* y = (float*)d_out;
  hipLaunchKernelGGL(stgcn_fused, dim3(Ndim * NCHUNK), dim3(256), 0, stream,
                     x, A, W, b, y);
}
